// MultiHeadSelfAttention_77936476553412
// MI455X (gfx1250) — matrix operand order repaired and checked by reading, not compiled, not run on hardware
//
#include <hip/hip_runtime.h>

// ---------------------------------------------------------------------------
// MHA forward for MI455X (gfx1250, wave32, WMMA 16x16x32 bf16 f32-accum,
// Tensor Data Mover staging of K/V and GEMM tiles into LDS).
// N=2, L=2048, E=1024, H=16, D=64
// ---------------------------------------------------------------------------

#define NB   2
#define LCTX 2048
#define EMB  1024
#define HEADS 16
#define HD   64

typedef __attribute__((ext_vector_type(16))) __bf16 bf16x16;
typedef __attribute__((ext_vector_type(8)))  float  f32x8;
typedef __attribute__((ext_vector_type(4)))  unsigned int u32x4;
typedef __attribute__((ext_vector_type(8)))  int    i32x8;
typedef __attribute__((ext_vector_type(4)))  int    i32x4;

__device__ __forceinline__ f32x8 wmma_bf16(bf16x16 a, bf16x16 b, f32x8 c) {
  return __builtin_amdgcn_wmma_f32_16x16x32_bf16(false, a, false, b, (short)0, c,
                                                 false, false);
}

// ---------------------------------------------------------------------------
// Tensor Data Mover: issue a (possibly padded) 2-D tile load Global -> LDS.
//   d0      : tile/tensor dim0 in 8-byte units (row segment length)
//   d1      : number of rows (tile/tensor dim1)
//   stride0 : row stride in 8-byte units
//   flags   : group1 word0 = data_size | pad_enable | pad_interval | pad_amount
// Descriptor layout per CDNA5 ISA.
// ---------------------------------------------------------------------------
#if __has_builtin(__builtin_amdgcn_tensor_load_to_lds)
#define USE_TDM 1
__device__ __forceinline__ void tdm_load_2d(void* lds, const void* gsrc,
                                            unsigned d0, unsigned d1,
                                            unsigned stride0, unsigned flags) {
  u32x4 g0;
  g0[0] = 1u;  // count=1 valid user descriptor, gather off
  g0[1] = (unsigned)(unsigned long long)(uintptr_t)lds;  // LDS byte address
  unsigned long long ga = (unsigned long long)(uintptr_t)gsrc;
  g0[2] = (unsigned)ga;                                  // global_addr[31:0]
  g0[3] = (unsigned)((ga >> 32) & 0x01FFFFFFull) | (2u << 30);  // addr[56:32]|type=2
  i32x8 g1;
  g1[0] = (int)flags;                          // wg_mask=0 | data_size | pad cfg
  g1[1] = (int)(d0 << 16);                     // tensor_dim0[15:0]
  g1[2] = (int)((d0 >> 16) | (d1 << 16));      // tensor_dim0[31:16] | tensor_dim1[15:0]
  g1[3] = (int)((d1 >> 16) | (d0 << 16));      // tensor_dim1[31:16] | tile_dim0
  g1[4] = (int)(d1 & 0xFFFFu);                 // tile_dim1 (tile_dim2 = 0)
  g1[5] = (int)stride0;                        // tensor_dim0_stride[31:0]
  g1[6] = 0;                                   // stride0 hi | stride1 lo
  g1[7] = 0;
  i32x4 zn = {};
#if defined(__clang_major__) && (__clang_major__ >= 23)
  i32x8 z8 = {};
  __builtin_amdgcn_tensor_load_to_lds(g0, g1, zn, zn, z8, 0);
#else
  __builtin_amdgcn_tensor_load_to_lds(g0, g1, zn, zn, 0);
#endif
}
// data_size=8B | pad_enable | pad 4 DW after every 32 DW (128B rows -> pitch 144B)
#define TDM_FLAGS_KV ((3u << 16) | (1u << 20) | (4u << 22) | (3u << 25))
// data_size=8B | pad_enable | pad 4 DW after every 16 DW (64B rows -> pitch 80B)
#define TDM_FLAGS_GEMM ((3u << 16) | (1u << 20) | (3u << 22) | (3u << 25))
#else
#define USE_TDM 0
#endif

// A-matrix fragment (16x32 bf16, ISA 7.12.2): lane (m = lane&15, half = lane>>4)
// element e -> K = k0 + 16*half + e (contiguous, matches B fragments)
__device__ __forceinline__ bf16x16 a_frag_f32(const float* __restrict__ row,
                                              int k0, int half) {
  bf16x16 a;
  const float* p0 = row + k0 + 16 * half;
  const float* p1 = p0 + 8;
#pragma unroll
  for (int e = 0; e < 8; ++e) { a[e] = (__bf16)p0[e]; a[e + 8] = (__bf16)p1[e]; }
  return a;
}

__device__ __forceinline__ bf16x16 a_frag_bf16(const __bf16* row, int k0, int half) {
  bf16x16 a;
  const __bf16* p0 = row + k0 + 16 * half;
  const __bf16* p1 = p0 + 8;
#pragma unroll
  for (int e = 0; e < 8; ++e) { a[e] = p0[e]; a[e + 8] = p1[e]; }
  return a;
}

// B fragment: lane (n = lane&15, half = lane>>4), element e -> K = k0+16*half+e.
__device__ __forceinline__ bf16x16 b_frag_contig(const __bf16* p) {
  bf16x16 b;
#pragma unroll
  for (int e = 0; e < 16; ++e) b[e] = p[e];
  return b;
}

__device__ __forceinline__ bf16x16 b_frag_strided(const __bf16* p, int pitch) {
  bf16x16 b;
#pragma unroll
  for (int e = 0; e < 16; ++e) b[e] = p[e * pitch];
  return b;
}

// ---------------------------------------------------------------------------
// Kernel 0: fp32 -> bf16 convert (for Wo)
// ---------------------------------------------------------------------------
__global__ __launch_bounds__(256) void cvt_bf16_kernel(const float* __restrict__ src,
                                                       __bf16* __restrict__ dst,
                                                       int n) {
  int i = (blockIdx.x * 256 + threadIdx.x) * 4;
  if (i + 3 < n) {
#pragma unroll
    for (int j = 0; j < 4; ++j) dst[i + j] = (__bf16)src[i + j];
  }
}

// ---------------------------------------------------------------------------
// Kernel 1: fused per-head projection  out[n,h,l,:] = x[n,l,h,:] @ W^T + b
// X: [N, L, E] fp32, W: [D, D] fp32, dst: [N*H, L, D] bf16 (head-major)
// grid: (L/128, N*H), block 256 (8 waves; 16 token rows per wave)
// ---------------------------------------------------------------------------
__global__ __launch_bounds__(256) void qkv_proj_kernel(const float* __restrict__ X,
                                                       const float* __restrict__ W,
                                                       const float* __restrict__ bias,
                                                       __bf16* __restrict__ dst) {
  __shared__ __bf16 Wl[64 * 72];
  const int tid = threadIdx.x;
#pragma unroll
  for (int i = 0; i < 16; ++i) {
    int idx = tid * 16 + i;
    Wl[(idx >> 6) * 72 + (idx & 63)] = (__bf16)W[idx];
  }
  __syncthreads();

  const int wave = tid >> 5, lane = tid & 31;
  const int m = lane & 15, half = lane >> 4;
  const int nh = blockIdx.y, n = nh >> 4, h = nh & 15;
  const int l0 = blockIdx.x * 128 + wave * 16;

  const float* row = X + ((size_t)n * LCTX + (l0 + m)) * EMB + h * HD;
  bf16x16 a0 = a_frag_f32(row, 0, half);
  bf16x16 a1 = a_frag_f32(row, 32, half);

  const size_t dbase = ((size_t)nh * LCTX + l0) * HD;
#pragma unroll
  for (int j = 0; j < 4; ++j) {
    f32x8 c = {};
    const __bf16* bp = Wl + (j * 16 + m) * 72 + 16 * half;
    c = wmma_bf16(a0, b_frag_contig(bp), c);
    c = wmma_bf16(a1, b_frag_contig(bp + 32), c);
    float bb = bias[j * 16 + m];
#pragma unroll
    for (int v = 0; v < 8; ++v)
      dst[dbase + (size_t)(v + 8 * half) * HD + j * 16 + m] = (__bf16)(c[v] + bb);
  }
}

// ---------------------------------------------------------------------------
// K/V tile staging: TDM issue (wave 0) or cooperative-copy fallback.
// ---------------------------------------------------------------------------
__device__ __forceinline__ void stage_kv(__bf16* kd, __bf16* vd,
                                         const __bf16* ks, const __bf16* vs,
                                         int key0, int tid) {
#if USE_TDM
  if (tid < 32) {
    // 8 KB contiguous, padded to 144B LDS pitch by the TDM
    tdm_load_2d(kd, ks + (size_t)key0 * HD, 1024u, 1u, 1024u, TDM_FLAGS_KV);
    tdm_load_2d(vd, vs + (size_t)key0 * HD, 1024u, 1u, 1024u, TDM_FLAGS_KV);
  }
#else
  const int r = tid >> 1, ch = tid & 1;
  const uint4* sk = (const uint4*)(ks + (size_t)(key0 + r) * HD + ch * 32);
  const uint4* sv = (const uint4*)(vs + (size_t)(key0 + r) * HD + ch * 32);
  uint4* dk = (uint4*)(kd + r * 72 + ch * 32);
  uint4* dv = (uint4*)(vd + r * 72 + ch * 32);
#pragma unroll
  for (int j = 0; j < 4; ++j) { dk[j] = sk[j]; dv[j] = sv[j]; }
#endif
}

// ---------------------------------------------------------------------------
// Kernel 2: flash attention per (n,h). Scale = 1/sqrt(E) = 1/32 (per reference).
// Double-buffered K/V tiles streamed by the TDM while WMMA computes.
// grid: (L/64, N*H), block 128 (4 waves x 16 query rows)
// ---------------------------------------------------------------------------
__global__ __launch_bounds__(128) void flash_attn_kernel(const __bf16* __restrict__ Qb,
                                                         const __bf16* __restrict__ Kb,
                                                         const __bf16* __restrict__ Vb,
                                                         const int* __restrict__ mask,
                                                         __bf16* __restrict__ Obuf) {
  __shared__ __bf16 Kt[2][64 * 72];
  __shared__ __bf16 Vt[2][64 * 72];
  __shared__ __bf16 Pt[4 * 16 * 72];

  const int tid = threadIdx.x, wave = tid >> 5, lane = tid & 31;
  const int m = lane & 15, half = lane >> 4;
  const int nh = blockIdx.y, n = nh >> 4, h = nh & 15;
  const int q0 = blockIdx.x * 64;
  const size_t kvbase = (size_t)nh * LCTX * HD;
  const __bf16* Kblk = Kb + kvbase;
  const __bf16* Vblk = Vb + kvbase;

  const __bf16* Qrow = Qb + kvbase + (size_t)(q0 + wave * 16 + m) * HD;
  bf16x16 aQ0 = a_frag_bf16(Qrow, 0, half);
  bf16x16 aQ1 = a_frag_bf16(Qrow, 32, half);

  f32x8 o[4] = {};
  float mrow[8], lrow[8];
#pragma unroll
  for (int v = 0; v < 8; ++v) { mrow[v] = -3.0e38f; lrow[v] = 0.f; }

  const int* mline = mask + n * LCTX;
  const float sc = 0.03125f;  // 1/sqrt(1024)
  const int NKB = LCTX / 64;

  stage_kv(Kt[0], Vt[0], Kblk, Vblk, 0, tid);  // prologue fill of buffer 0

  for (int kb = 0; kb < NKB; ++kb) {
#if USE_TDM
    if (tid < 32) __builtin_amdgcn_s_wait_tensorcnt(0);  // tile for kb landed
#endif
    __syncthreads();
    const int cur = kb & 1;
    if (kb + 1 < NKB)  // overlap next tile DMA with this block's WMMA work
      stage_kv(Kt[cur ^ 1], Vt[cur ^ 1], Kblk, Vblk, (kb + 1) * 64, tid);

    const __bf16* Kc = Kt[cur];
    const __bf16* Vc = Vt[cur];
    const int key0 = kb * 64;

    // S = Q @ K^T : 4 key-tiles of 16, K-dim = 64 features (2 wmma each)
    f32x8 s[4];
#pragma unroll
    for (int t = 0; t < 4; ++t) {
      f32x8 z = {};
      const __bf16* bp = Kc + (t * 16 + m) * 72 + 16 * half;  // B[d][key]=Kt[key][d]
      z = wmma_bf16(aQ0, b_frag_contig(bp), z);
      z = wmma_bf16(aQ1, b_frag_contig(bp + 32), z);
      s[t] = z;
    }
    // scale + mask (reference: masked -> -1e-30 before softmax)
#pragma unroll
    for (int t = 0; t < 4; ++t) {
      const bool msk = (mline[key0 + t * 16 + m] == 0);
#pragma unroll
      for (int v = 0; v < 8; ++v) {
        float x = s[t][v] * sc;
        s[t][v] = msk ? -1e-30f : x;
      }
    }
    // online softmax: row v+8*half lives across 16-lane half-groups
#pragma unroll
    for (int v = 0; v < 8; ++v) {
      float mx = fmaxf(fmaxf(s[0][v], s[1][v]), fmaxf(s[2][v], s[3][v]));
      mx = fmaxf(mx, __shfl_xor(mx, 1, 32));
      mx = fmaxf(mx, __shfl_xor(mx, 2, 32));
      mx = fmaxf(mx, __shfl_xor(mx, 4, 32));
      mx = fmaxf(mx, __shfl_xor(mx, 8, 32));
      const float nm = fmaxf(mrow[v], mx);
      const float corr = __expf(mrow[v] - nm);
      mrow[v] = nm;
      float rs = 0.f;
#pragma unroll
      for (int t = 0; t < 4; ++t) {
        float p = __expf(s[t][v] - nm);
        s[t][v] = p;
        rs += p;
      }
      rs += __shfl_xor(rs, 1, 32);
      rs += __shfl_xor(rs, 2, 32);
      rs += __shfl_xor(rs, 4, 32);
      rs += __shfl_xor(rs, 8, 32);
      lrow[v] = lrow[v] * corr + rs;
#pragma unroll
      for (int t = 0; t < 4; ++t) o[t][v] *= corr;
    }
    // relayout P (C layout -> A layout) through per-wave LDS strip
    __bf16* Pw = Pt + wave * (16 * 72);
#pragma unroll
    for (int t = 0; t < 4; ++t)
#pragma unroll
      for (int v = 0; v < 8; ++v)
        Pw[(v + 8 * half) * 72 + t * 16 + m] = (__bf16)s[t][v];
    bf16x16 aP0 = a_frag_bf16(Pw + m * 72, 0, half);
    bf16x16 aP1 = a_frag_bf16(Pw + m * 72, 32, half);
    // O += P @ V  (B[key][d] strided)
#pragma unroll
    for (int j = 0; j < 4; ++j) {
      o[j] = wmma_bf16(aP0, b_frag_strided(Vc + (16 * half) * 72 + j * 16 + m, 72), o[j]);
      o[j] = wmma_bf16(aP1, b_frag_strided(Vc + (32 + 16 * half) * 72 + j * 16 + m, 72), o[j]);
    }
  }

  // epilogue: normalize and scatter back to interleaved [N, L, E] bf16
#pragma unroll
  for (int v = 0; v < 8; ++v) {
    const float inv = 1.0f / lrow[v];
    const int q = q0 + wave * 16 + v + 8 * half;
    const size_t ob = ((size_t)n * LCTX + q) * EMB + h * HD;
#pragma unroll
    for (int t = 0; t < 4; ++t)
      Obuf[ob + t * 16 + m] = (__bf16)(o[t][v] * inv);
  }
}

// ---------------------------------------------------------------------------
// Kernel 3: output projection  out = X @ Wo^T + bo
// X: [4096, 1024] bf16, Wob: [1024, 1024] bf16, out fp32.
// Tiles staged by TDM 2-D descriptors (64B row segments, padded to 80B pitch).
// grid: (M/64, E/256), block 256 (8 waves = 4 row-tiles x 2 col-groups)
// ---------------------------------------------------------------------------
__global__ __launch_bounds__(256) void out_proj_kernel(const __bf16* __restrict__ Xb,
                                                       const __bf16* __restrict__ Wb,
                                                       const float* __restrict__ bo,
                                                       float* __restrict__ out) {
  __shared__ __bf16 Xt[64 * 40];    // 64 rows x 32 k, pitch 40
  __shared__ __bf16 Wt[256 * 40];   // 256 cols x 32 k, pitch 40
  const int tid = threadIdx.x, wave = tid >> 5, lane = tid & 31;
  const int m = lane & 15, half = lane >> 4;
  const int wm = wave & 3, wn = wave >> 2;
  const int m0 = blockIdx.x * 64;
  const int nc0 = blockIdx.y * 256;

  f32x8 acc[8] = {};
  for (int kc = 0; kc < EMB / 32; ++kc) {
    __syncthreads();
#if USE_TDM
    if (tid < 32) {
      // 64B-wide row segments, global row stride 2048B (=256 x 8B units)
      tdm_load_2d(Xt, Xb + (size_t)m0 * EMB + kc * 32, 8u, 64u, 256u, TDM_FLAGS_GEMM);
      tdm_load_2d(Wt, Wb + (size_t)nc0 * EMB + kc * 32, 8u, 256u, 256u, TDM_FLAGS_GEMM);
      __builtin_amdgcn_s_wait_tensorcnt(0);
    }
#else
    {
      const int r = tid >> 2, ch = tid & 3;
      *(uint4*)(Xt + r * 40 + ch * 8) =
          *(const uint4*)(Xb + (size_t)(m0 + r) * EMB + kc * 32 + ch * 8);
      const uint4* s = (const uint4*)(Wb + (size_t)(nc0 + tid) * EMB + kc * 32);
      uint4* d = (uint4*)(Wt + tid * 40);
#pragma unroll
      for (int j = 0; j < 4; ++j) d[j] = s[j];
    }
#endif
    __syncthreads();
    bf16x16 a = a_frag_bf16(Xt + (wm * 16 + m) * 40, 0, half);
#pragma unroll
    for (int j = 0; j < 8; ++j) {
      const int ncl = wn * 128 + j * 16 + m;
      acc[j] = wmma_bf16(a, b_frag_contig(Wt + ncl * 40 + 16 * half), acc[j]);
    }
  }
#pragma unroll
  for (int j = 0; j < 8; ++j) {
    const int col = nc0 + wn * 128 + j * 16 + m;
    const float bb = bo[col];
#pragma unroll
    for (int v = 0; v < 8; ++v) {
      const int row = m0 + wm * 16 + v + 8 * half;
      out[(size_t)row * EMB + col] = acc[j][v] + bb;
    }
  }
}

// ---------------------------------------------------------------------------
extern "C" void kernel_launch(void* const* d_in, const int* in_sizes, int n_in,
                              void* d_out, int out_size, void* d_ws, size_t ws_size,
                              hipStream_t stream) {
  const float* key   = (const float*)d_in[0];
  const float* query = (const float*)d_in[1];
  const float* value = (const float*)d_in[2];
  const int*   mask  = (const int*)d_in[3];
  const float* Wk = (const float*)d_in[4];
  const float* bk = (const float*)d_in[5];
  const float* Wq = (const float*)d_in[6];
  const float* bq = (const float*)d_in[7];
  const float* Wv = (const float*)d_in[8];
  const float* bv = (const float*)d_in[9];
  const float* Wo = (const float*)d_in[10];
  const float* bo = (const float*)d_in[11];
  float* out = (float*)d_out;

  const size_t NHLD = (size_t)NB * HEADS * LCTX * HD;  // 4M elems
  __bf16* Qb  = (__bf16*)d_ws;
  __bf16* Kb  = Qb + NHLD;
  __bf16* Vb  = Kb + NHLD;
  __bf16* Ob  = Vb + NHLD;                     // [N, L, E] bf16
  __bf16* Wob = Ob + (size_t)NB * LCTX * EMB;  // [E, E] bf16

  cvt_bf16_kernel<<<EMB * EMB / 1024, 256, 0, stream>>>(Wo, Wob, EMB * EMB);

  dim3 gp(LCTX / 128, NB * HEADS);
  qkv_proj_kernel<<<gp, 256, 0, stream>>>(query, Wq, bq, Qb);
  qkv_proj_kernel<<<gp, 256, 0, stream>>>(key,   Wk, bk, Kb);
  qkv_proj_kernel<<<gp, 256, 0, stream>>>(value, Wv, bv, Vb);

  flash_attn_kernel<<<dim3(LCTX / 64, NB * HEADS), 128, 0, stream>>>(Qb, Kb, Vb, mask, Ob);

  out_proj_kernel<<<dim3(NB * LCTX / 64, EMB / 256), 256, 0, stream>>>(Ob, Wob, bo, out);
}
